// PosUpdate_12017318494547
// MI455X (gfx1250) — compile-verified, matrix operand order checked by reading, not run on hardware
//
#include <hip/hip_runtime.h>

typedef __attribute__((ext_vector_type(16))) __bf16 v16bf;
typedef __attribute__((ext_vector_type(8)))  float  v8f;

// ---------- helpers ----------

__device__ __forceinline__ unsigned short f2bfbits(float f) {
  union { float f; unsigned u; } v; v.f = f;
  unsigned r = v.u + 0x7fffu + ((v.u >> 16) & 1u);   // round-to-nearest-even
  return (unsigned short)(r >> 16);
}

__device__ __forceinline__ unsigned pack2bf(float a, float b) {
  return (unsigned)f2bfbits(a) | ((unsigned)f2bfbits(b) << 16);
}

__device__ __forceinline__ v8f wmma_bf16(v16bf a, v16bf b, v8f c) {
  return __builtin_amdgcn_wmma_f32_16x16x32_bf16(false, a, false, b, (short)0, c, false, false);
}

// A fragment (16x32 bf16) from LDS row-major ushort array, row stride ld.
__device__ __forceinline__ v16bf load_frag_a_lds(const unsigned short* s, int ld, int kbase, int lane) {
  int m  = lane & 15;
  int kh = (lane >> 4) << 3;
  const unsigned short* row = s + m * ld + kbase + kh;
  union { v16bf bf; unsigned u[8]; } fr;
#pragma unroll
  for (int v = 0; v < 8; ++v) {
    int ks = ((v & 4) << 2) + ((v & 3) << 1);  // +16 for v>=4
    fr.u[v] = (unsigned)row[ks] | ((unsigned)row[ks + 1] << 16);
  }
  return fr.bf;
}

// B fragment from pre-packed bf16 buffer: per (ktile,ntile), 32 lanes x 8 dwords contiguous.
__device__ __forceinline__ v16bf load_frag_b_packed(const unsigned* __restrict__ P, int NT,
                                                    int kt, int nt, int lane) {
  const unsigned* p = P + ((((size_t)kt * NT + nt) << 5) + lane) * 8;
  union { v16bf bf; unsigned u[8]; } fr;
#pragma unroll
  for (int v = 0; v < 8; ++v) fr.u[v] = p[v];   // -> two global_load_b128
  return fr.bf;
}

// ---------- kernels ----------

__global__ void zero_kernel(float* __restrict__ p, int n) {
  int i = blockIdx.x * 256 + threadIdx.x;
  if (i < n) p[i] = 0.f;
}

// Pack f32 B[K][N] (row-major) into bf16 fragment layout. One thread per output dword.
__global__ void pack_b_kernel(const float* __restrict__ B, unsigned* __restrict__ P, int K, int N) {
  int d = blockIdx.x * 256 + threadIdx.x;
  if (d >= (K * N) >> 1) return;
  int v    = d & 7;
  int lane = (d >> 3) & 31;
  int t    = d >> 8;              // kt*NT + nt
  int NT   = N >> 4;
  int kt = t / NT, nt = t - kt * NT;
  int k = kt * 32 + ((lane >> 4) << 3) + ((v & 4) << 2) + ((v & 3) << 1);
  int n = nt * 16 + (lane & 15);
  P[d] = pack2bf(B[(size_t)k * N + n], B[(size_t)(k + 1) * N + n]);
}

// C[M,N] = A[M,256] @ B[256,N] + bias ; 4 waves/block, block tile 16 x 64.
__global__ void __launch_bounds__(128)
gemm_bias_k256(const float* __restrict__ A, const unsigned* __restrict__ Bp,
               const float* __restrict__ bias, float* __restrict__ C, int N) {
  __shared__ alignas(16) unsigned short sA[16 * 256];
  const int K = 256;
  int tid = threadIdx.x, lane = tid & 31, wave = tid >> 5;
  size_t m0 = (size_t)blockIdx.x * 16;
  int n0 = blockIdx.y * 64 + wave * 16;
  int NT = N >> 4, nt = n0 >> 4;

  for (int i = tid; i < (16 * K) >> 2; i += 128) {      // float4 staging
    int r = (i << 2) >> 8, c = (i << 2) & 255;
    float4 x = *(const float4*)(A + (m0 + r) * K + c);
    unsigned* dst = (unsigned*)&sA[(r << 8) + c];
    dst[0] = pack2bf(x.x, x.y);
    dst[1] = pack2bf(x.z, x.w);
  }
  __syncthreads();

  v8f acc = {};
#pragma unroll
  for (int kb = 0; kb < K; kb += 32) {
    v16bf a = load_frag_a_lds(sA, K, kb, lane);
    v16bf b = load_frag_b_packed(Bp, NT, kb >> 5, nt, lane);
    acc = wmma_bf16(a, b, acc);
  }
  int n = n0 + (lane & 15);
  float bv = bias[n];
#pragma unroll
  for (int v = 0; v < 8; ++v) {
    int m = v + ((lane >> 4) << 3);
    C[(m0 + m) * (size_t)N + n] = acc[v] + bv;
  }
}

// rowwise LayerNorm + ReLU in place, dim == 256, one block (256 thr) per row
__global__ void __launch_bounds__(256)
ln_relu_256(float* __restrict__ X, const float* __restrict__ g, const float* __restrict__ be) {
  __shared__ float sS[8], sS2[8];
  int tid = threadIdx.x;
  size_t row = blockIdx.x;
  float x = X[row * 256 + tid];
  float s = x, ss = x * x;
#pragma unroll
  for (int off = 16; off; off >>= 1) {
    s  += __shfl_xor(s, off, 32);
    ss += __shfl_xor(ss, off, 32);
  }
  if ((tid & 31) == 0) { sS[tid >> 5] = s; sS2[tid >> 5] = ss; }
  __syncthreads();
  float tS = 0.f, tS2 = 0.f;
#pragma unroll
  for (int i = 0; i < 8; ++i) { tS += sS[i]; tS2 += sS2[i]; }
  float mu  = tS * (1.f / 256.f);
  float var = tS2 * (1.f / 256.f) - mu * mu;
  float y = (x - mu) * rsqrtf(var + 1e-5f) * g[tid] + be[tid];
  X[row * 256 + tid] = fmaxf(y, 0.f);
}

// Fully fused edge kernel: 16 edges / block (128 thr = 4 waves).
//   node_in = Lout[ei_l] * Rout[ei_r]  (gathered, L2-resident tables) -> LDS bf16
//   prod  = (h_edge@bondW) * (node_in@nodeW)            [LDS]
//   hid   = prod @ iW1 + ib1 ; inter = relu(LN).iW2+ib2
//   ghid  = concat(h_edge,node_in,time) @ gW1 + gb1 ; gate = relu(LN).gW2+gb2
//   force = inter*sigmoid(gate)*rel/(d*(d+1)) -> atomic scatter on ei_left
__global__ void __launch_bounds__(128)
edge_fused_kernel(const float* __restrict__ h_edge,
                  const float* __restrict__ Lout, const float* __restrict__ Rout,
                  const unsigned* __restrict__ bondWp, const unsigned* __restrict__ nodeWp,
                  const unsigned* __restrict__ iW1p, const float* __restrict__ ib1,
                  const float* __restrict__ ig1, const float* __restrict__ ibe1,
                  const float* __restrict__ iW2, const float* __restrict__ ib2,
                  const unsigned* __restrict__ gW1p, const float* __restrict__ gW1,
                  const float* __restrict__ gb1, const float* __restrict__ gg1,
                  const float* __restrict__ gbe1, const float* __restrict__ gW2,
                  const float* __restrict__ gb2,
                  const float* __restrict__ etime, const int* __restrict__ ei,
                  const float* __restrict__ rel, const float* __restrict__ dist,
                  float* __restrict__ out, int nE) {
  __shared__ alignas(16) unsigned short sEdge[16 * 128];
  __shared__ alignas(16) unsigned short sNin[16 * 128];
  __shared__ alignas(16) unsigned short sProd[16 * 256];
  __shared__ float sHid[16 * 256];
  __shared__ float sGH[16 * 32];
  __shared__ float sRed[16 * 8], sRed2[16 * 8];
  __shared__ float sMu[16], sRs[16], sT[16], sGate[16];
  __shared__ int   sIL[16], sIR[16];

  int tid = threadIdx.x, lane = tid & 31, wave = tid >> 5;
  size_t e0 = (size_t)blockIdx.x * 16;

  if (tid < 16) {
    sIL[tid] = ei[e0 + tid];
    sIR[tid] = ei[(size_t)nE + e0 + tid];
    sT[tid]  = etime[e0 + tid];
  }
  // h_edge staging (does not need indices)
  for (int i = tid; i < (16 * 128) >> 2; i += 128) {
    int r = (i << 2) >> 7, c = (i << 2) & 127;
    float4 x = *(const float4*)(h_edge + (e0 + r) * 128 + c);
    unsigned* de = (unsigned*)&sEdge[(r << 7) + c];
    de[0] = pack2bf(x.x, x.y); de[1] = pack2bf(x.z, x.w);
  }
  __syncthreads();

  // gathered node_in staging: 16 rows x 32 float4 (rows are 512B contiguous)
  for (int i = tid; i < 16 * 32; i += 128) {
    int r = i >> 5, c4 = i & 31;
    size_t l = (size_t)sIL[r], rr = (size_t)sIR[r];
    float4 a = ((const float4*)Lout)[l * 32 + c4];
    float4 b = ((const float4*)Rout)[rr * 32 + c4];
    unsigned* dn = (unsigned*)&sNin[(r << 7) + (c4 << 2)];
    dn[0] = pack2bf(a.x * b.x, a.y * b.y);
    dn[1] = pack2bf(a.z * b.z, a.w * b.w);
  }
  __syncthreads();

  // dual GEMM (K=128) + elementwise product -> sProd (bf16)
#pragma unroll
  for (int i = 0; i < 4; ++i) {
    int nt = wave * 4 + i, nb = nt * 16;
    v8f acc1 = {}, acc2 = {};
#pragma unroll
    for (int kb = 0; kb < 128; kb += 32) {
      v16bf a1 = load_frag_a_lds(sEdge, 128, kb, lane);
      v16bf bb = load_frag_b_packed(bondWp, 16, kb >> 5, nt, lane);
      acc1 = wmma_bf16(a1, bb, acc1);
      v16bf a2 = load_frag_a_lds(sNin, 128, kb, lane);
      v16bf bn = load_frag_b_packed(nodeWp, 16, kb >> 5, nt, lane);
      acc2 = wmma_bf16(a2, bn, acc2);
    }
    int n = nb + (lane & 15);
#pragma unroll
    for (int v = 0; v < 8; ++v) {
      int m = v + ((lane >> 4) << 3);
      sProd[m * 256 + n] = f2bfbits(acc1[v] * acc2[v]);
    }
  }

  // gate hidden GEMM (16x257 @ 257x32): waves 0,1 each one 16-col tile
  if (wave < 2) {
    int nt = wave;
    v8f acc = {};
#pragma unroll
    for (int kb = 0; kb < 128; kb += 32) {
      v16bf a = load_frag_a_lds(sEdge, 128, kb, lane);
      v16bf b = load_frag_b_packed(gW1p, 2, kb >> 5, nt, lane);
      acc = wmma_bf16(a, b, acc);
    }
#pragma unroll
    for (int kb = 0; kb < 128; kb += 32) {
      v16bf a = load_frag_a_lds(sNin, 128, kb, lane);
      v16bf b = load_frag_b_packed(gW1p, 2, 4 + (kb >> 5), nt, lane);
      acc = wmma_bf16(a, b, acc);
    }
    int n = nt * 16 + (lane & 15);
    float wlast = gW1[256 * 32 + n];   // time row (k = 256)
    float bv = gb1[n];
#pragma unroll
    for (int v = 0; v < 8; ++v) {
      int m = v + ((lane >> 4) << 3);
      sGH[m * 32 + n] = acc[v] + sT[m] * wlast + bv;
    }
  }
  __syncthreads();

  // inter hidden GEMM (K=256) -> sHid (f32)
#pragma unroll
  for (int i = 0; i < 4; ++i) {
    int nt = wave * 4 + i, nb = nt * 16;
    v8f acc = {};
#pragma unroll
    for (int kb = 0; kb < 256; kb += 32) {
      v16bf a = load_frag_a_lds(sProd, 256, kb, lane);
      v16bf b = load_frag_b_packed(iW1p, 16, kb >> 5, nt, lane);
      acc = wmma_bf16(a, b, acc);
    }
    int n = nb + (lane & 15);
    float bv = ib1[n];
#pragma unroll
    for (int v = 0; v < 8; ++v) {
      int m = v + ((lane >> 4) << 3);
      sHid[m * 256 + n] = acc[v] + bv;
    }
  }
  __syncthreads();

  // inter LN partials: 8 threads per row, 32 elems each
  {
    int r = tid >> 3, ch = tid & 7;
    float s = 0.f, ss = 0.f;
    for (int j = ch * 32; j < ch * 32 + 32; ++j) {
      float x = sHid[r * 256 + j];
      s += x; ss += x * x;
    }
    sRed[r * 8 + ch] = s; sRed2[r * 8 + ch] = ss;
  }
  __syncthreads();
  if (tid < 16) {
    float s = 0.f, ss = 0.f;
#pragma unroll
    for (int c = 0; c < 8; ++c) { s += sRed[tid * 8 + c]; ss += sRed2[tid * 8 + c]; }
    float mu  = s * (1.f / 256.f);
    float var = ss * (1.f / 256.f) - mu * mu;
    sMu[tid] = mu; sRs[tid] = rsqrtf(var + 1e-5f);

    // gate LN + dot for row tid (32 elements, cheap)
    float gs = 0.f, gss = 0.f;
#pragma unroll
    for (int j = 0; j < 32; ++j) { float x = sGH[tid * 32 + j]; gs += x; gss += x * x; }
    float gmu = gs * (1.f / 32.f);
    float gvr = gss * (1.f / 32.f) - gmu * gmu;
    float grs = rsqrtf(gvr + 1e-5f);
    float gd = 0.f;
#pragma unroll
    for (int j = 0; j < 32; ++j) {
      float y = (sGH[tid * 32 + j] - gmu) * grs * gg1[j] + gbe1[j];
      gd += fmaxf(y, 0.f) * gW2[j];
    }
    sGate[tid] = gd + gb2[0];
  }
  __syncthreads();

  // inter dot partials
  {
    int r = tid >> 3, ch = tid & 7;
    float mu = sMu[r], rs = sRs[r];
    float s = 0.f;
    for (int j = ch * 32; j < ch * 32 + 32; ++j) {
      float y = (sHid[r * 256 + j] - mu) * rs * ig1[j] + ibe1[j];
      s += fmaxf(y, 0.f) * iW2[j];
    }
    sRed[r * 8 + ch] = s;
  }
  __syncthreads();
  if (tid < 16) {
    float s = 0.f;
#pragma unroll
    for (int c = 0; c < 8; ++c) s += sRed[tid * 8 + c];
    float inter_r = s + ib2[0];
    float gate = sGate[tid];
    float w = inter_r * (1.f / (1.f + __expf(-gate)));
    size_t e = e0 + tid;
    float d = dist[e];
    float f = w / (d * (d + 1.f));
    size_t dst = (size_t)sIL[tid];        // ei_left
    atomicAdd(&out[dst * 3 + 0], f * rel[e * 3 + 0]);
    atomicAdd(&out[dst * 3 + 1], f * rel[e * 3 + 1]);
    atomicAdd(&out[dst * 3 + 2], f * rel[e * 3 + 2]);
  }
}

// ---------- launch ----------

extern "C" void kernel_launch(void* const* d_in, const int* in_sizes, int n_in,
                              void* d_out, int out_size, void* d_ws, size_t ws_size,
                              hipStream_t stream) {
  (void)in_sizes; (void)n_in; (void)out_size; (void)ws_size;

  const float* h_node = (const float*)d_in[0];
  const float* h_edge = (const float*)d_in[1];
  const int*   ei     = (const int*)d_in[2];
  const float* rel    = (const float*)d_in[3];
  const float* dist   = (const float*)d_in[4];
  const float* etime  = (const float*)d_in[5];
  const float* bond_W = (const float*)d_in[6];
  const float* gW1  = (const float*)d_in[7];
  const float* gW2  = (const float*)d_in[8];
  const float* gb1  = (const float*)d_in[9];
  const float* gb2  = (const float*)d_in[10];
  const float* gbe1 = (const float*)d_in[11];
  const float* gg1  = (const float*)d_in[12];
  const float* iW1  = (const float*)d_in[13];
  const float* iW2  = (const float*)d_in[14];
  const float* ib1  = (const float*)d_in[15];
  const float* ib2  = (const float*)d_in[16];
  const float* ibe1 = (const float*)d_in[17];
  const float* ig1  = (const float*)d_in[18];
  const float* lW1  = (const float*)d_in[19];
  const float* lW2  = (const float*)d_in[20];
  const float* lb1  = (const float*)d_in[21];
  const float* lb2  = (const float*)d_in[22];
  const float* lbe1 = (const float*)d_in[23];
  const float* lg1  = (const float*)d_in[24];
  const float* node_W = (const float*)d_in[25];
  const float* rW1  = (const float*)d_in[26];
  const float* rW2  = (const float*)d_in[27];
  const float* rb1  = (const float*)d_in[28];
  const float* rb2  = (const float*)d_in[29];
  const float* rbe1 = (const float*)d_in[30];
  const float* rg1  = (const float*)d_in[31];

  const int nN = 50000, nE = 500000;

  // packed-weight region (dwords), then float buffers
  unsigned* lW1p  = (unsigned*)d_ws;              // 256*256/2 = 32768 dw
  unsigned* lW2p  = lW1p  + 32768;                // 256*128/2 = 16384
  unsigned* rW1p  = lW2p  + 16384;                // 32768
  unsigned* rW2p  = rW1p  + 32768;                // 16384
  unsigned* bondp = rW2p  + 16384;                // 128*256/2 = 16384
  unsigned* nodep = bondp + 16384;                // 16384
  unsigned* iW1p  = nodep + 16384;                // 32768
  unsigned* gW1p  = iW1p  + 32768;                // 256*32/2  = 4096
  float* fbase = (float*)(gW1p + 4096);
  float* Lout  = fbase;                           // nN*128
  float* Rout  = Lout + (size_t)nN * 128;         // nN*128
  float* Hid   = Rout + (size_t)nN * 128;         // nN*256 (scratch, reused)

  float* out = (float*)d_out;

  zero_kernel<<<(nN * 3 + 255) / 256, 256, 0, stream>>>(out, nN * 3);

  // pack weights to bf16 fragment layout (one-time per launch, tiny)
  pack_b_kernel<<<128, 256, 0, stream>>>(lW1, lW1p, 256, 256);
  pack_b_kernel<<<64, 256, 0, stream>>>(lW2, lW2p, 256, 128);
  pack_b_kernel<<<128, 256, 0, stream>>>(rW1, rW1p, 256, 256);
  pack_b_kernel<<<64, 256, 0, stream>>>(rW2, rW2p, 256, 128);
  pack_b_kernel<<<64, 256, 0, stream>>>(bond_W, bondp, 128, 256);
  pack_b_kernel<<<64, 256, 0, stream>>>(node_W, nodep, 128, 256);
  pack_b_kernel<<<128, 256, 0, stream>>>(iW1, iW1p, 256, 256);
  pack_b_kernel<<<16, 256, 0, stream>>>(gW1, gW1p, 256, 32);   // first 256 rows

  // left node MLP
  gemm_bias_k256<<<dim3(nN / 16, 4), 128, 0, stream>>>(h_node, lW1p, lb1, Hid, 256);
  ln_relu_256<<<nN, 256, 0, stream>>>(Hid, lg1, lbe1);
  gemm_bias_k256<<<dim3(nN / 16, 2), 128, 0, stream>>>(Hid, lW2p, lb2, Lout, 128);
  // right node MLP
  gemm_bias_k256<<<dim3(nN / 16, 4), 128, 0, stream>>>(h_node, rW1p, rb1, Hid, 256);
  ln_relu_256<<<nN, 256, 0, stream>>>(Hid, rg1, rbe1);
  gemm_bias_k256<<<dim3(nN / 16, 2), 128, 0, stream>>>(Hid, rW2p, rb2, Rout, 128);
  // fully fused edge pipeline (gather + inter + gate + force + scatter)
  edge_fused_kernel<<<nE / 16, 128, 0, stream>>>(h_edge, Lout, Rout, bondp, nodep,
                                                 iW1p, ib1, ig1, ibe1, iW2, ib2,
                                                 gW1p, gW1, gb1, gg1, gbe1, gW2, gb2,
                                                 etime, ei, rel, dist, out, nE);
}
